// VideoSummarizationModel_3599182594077
// MI455X (gfx1250) — compile-verified, matrix operand order checked by reading, not run on hardware
//
#include <hip/hip_runtime.h>
#include <hip/hip_bf16.h>

// ---------------------------------------------------------------------------
// Shapes from the reference
//   T=2048, W=32, D=512, HEADS=8, H=512, S=16, L=128, A=1024, DH=64
//   M = T*W = S*W*L = 65536 for all big GEMMs.
// ---------------------------------------------------------------------------

typedef __attribute__((ext_vector_type(16))) _Float16 v16h;
typedef __attribute__((ext_vector_type(8)))  _Float16 v8h;
typedef __attribute__((ext_vector_type(8)))  float    v8f;

union V16U { v16h v; v8h h[2]; };

#define TILE_M 128
#define TILE_N 128
#define TILE_K 32
#define LDSS   40   // LDS row stride in halves (32 + 8 pad)

// ---------------------------------------------------------------------------
// Tiled WMMA GEMM:  C[M,N] = f16( relu?( A[M,K](f16) @ Bt[N,K]^T + bias[N] ) )
// Bt is B stored transposed ([N rows, K cols]) so A and B fragments load the
// same way.  Block = 256 threads = 8 waves; wave (wm in 0..3, wn in 0..1)
// computes a 32x64 sub-tile = 2x4 grid of 16x16 WMMA tiles.
// M,N multiples of 128; K multiple of 32.
// ---------------------------------------------------------------------------
__global__ __launch_bounds__(256) void gemm_wmma_f16(
    const _Float16* __restrict__ A, const _Float16* __restrict__ Bt,
    const float* __restrict__ bias, _Float16* __restrict__ C,
    int M, int N, int K, int relu)
{
    __shared__ alignas(16) _Float16 As[TILE_M * LDSS];
    __shared__ alignas(16) _Float16 Bs[TILE_N * LDSS];

    const int tid  = threadIdx.x;
    const int m0   = blockIdx.y * TILE_M;
    const int n0   = blockIdx.x * TILE_N;
    const int wave = tid >> 5, lane = tid & 31;
    const int wm   = wave >> 1, wn = wave & 1;
    const int lr   = lane & 15, hi = lane >> 4;
    const int lrow = tid >> 1;              // 0..127: row loaded by this thread
    const int lkc  = (tid & 1) * 16;        // 0 or 16 halves within the K-tile

    v8f zero = {};
    v8f acc[2][4];
#pragma unroll
    for (int i = 0; i < 2; ++i)
#pragma unroll
        for (int j = 0; j < 4; ++j) acc[i][j] = zero;

    const int nk = K >> 5;
    for (int kt = 0; kt < nk; ++kt) {
        const int k0 = kt << 5;
        {   // global -> LDS (16 halves per thread per matrix)
            const v8h* ga = (const v8h*)(A  + (size_t)(m0 + lrow) * K + k0 + lkc);
            const v8h* gb = (const v8h*)(Bt + (size_t)(n0 + lrow) * K + k0 + lkc);
            v8h* la = (v8h*)(As + lrow * LDSS + lkc);
            v8h* lb = (v8h*)(Bs + lrow * LDSS + lkc);
            la[0] = ga[0]; la[1] = ga[1];
            lb[0] = gb[0]; lb[1] = gb[1];
            if (kt + 1 < nk) {   // gfx1250 global_prefetch_b8 of next K-tile
                __builtin_prefetch(A  + (size_t)(m0 + lrow) * K + k0 + 32 + lkc, 0, 1);
                __builtin_prefetch(Bt + (size_t)(n0 + lrow) * K + k0 + 32 + lkc, 0, 1);
            }
        }
        __syncthreads();

        v16h af[2], bf[4];
#pragma unroll
        for (int mi = 0; mi < 2; ++mi) {
            const int row = wm * 32 + mi * 16 + lr;
            V16U t;
            t.h[0] = *(const v8h*)(As + row * LDSS + hi * 8);        // K 0..7 / 8..15
            t.h[1] = *(const v8h*)(As + row * LDSS + 16 + hi * 8);   // K 16..23 / 24..31
            af[mi] = t.v;
        }
#pragma unroll
        for (int ni = 0; ni < 4; ++ni) {
            const int row = wn * 64 + ni * 16 + lr;
            V16U t;
            t.h[0] = *(const v8h*)(Bs + row * LDSS + hi * 8);
            t.h[1] = *(const v8h*)(Bs + row * LDSS + 16 + hi * 8);
            bf[ni] = t.v;
        }
#pragma unroll
        for (int mi = 0; mi < 2; ++mi)
#pragma unroll
            for (int ni = 0; ni < 4; ++ni)
                acc[mi][ni] = __builtin_amdgcn_wmma_f32_16x16x32_f16(
                    false, af[mi], false, bf[ni], (short)0, acc[mi][ni], false, false);
        __syncthreads();
    }

    // epilogue: C/D layout: VGPR r -> (M = hi*8 + r, N = lr) within each 16x16 tile
#pragma unroll
    for (int mi = 0; mi < 2; ++mi) {
#pragma unroll
        for (int ni = 0; ni < 4; ++ni) {
            const int n  = n0 + wn * 64 + ni * 16 + lr;
            const float bv = bias ? bias[n] : 0.0f;
#pragma unroll
            for (int r = 0; r < 8; ++r) {
                const int m = m0 + wm * 32 + mi * 16 + hi * 8 + r;
                float v = acc[mi][ni][r] + bv;
                if (relu) v = fmaxf(v, 0.0f);
                C[(size_t)m * N + n] = (_Float16)v;
            }
        }
    }
}

// --------------------------- small helper kernels ---------------------------

__global__ void f32_to_f16(const float* __restrict__ src, _Float16* __restrict__ dst, size_t n)
{
    size_t i = (size_t)blockIdx.x * blockDim.x + threadIdx.x;
    if (i < n) dst[i] = (_Float16)src[i];
}

// W[K,N] f32 -> Wt[N,K] f16
__global__ void wprep_kernel(const float* __restrict__ src, _Float16* __restrict__ dst, int K, int N)
{
    int idx = blockIdx.x * blockDim.x + threadIdx.x;
    if (idx >= K * N) return;
    int n = idx / K, k = idx % K;
    dst[idx] = (_Float16)src[k * N + n];
}

// src[R,C] f16 -> dst[C,R] f16
__global__ void transpose_f16(const _Float16* __restrict__ src, _Float16* __restrict__ dst, int R, int Ccols)
{
    int idx = blockIdx.x * blockDim.x + threadIdx.x;
    if (idx >= R * Ccols) return;
    int r = idx / Ccols, c = idx % Ccols;
    dst[(size_t)c * R + r] = src[idx];
}

// Frame-level attention over the 8 head-slots per token; DH=64, scale 1/8.
// grid = 65536 tokens; block = 256 (wave per head); lane handles dims {l, l+32}.
__global__ __launch_bounds__(256) void frame_attn(
    const _Float16* __restrict__ Q, const _Float16* __restrict__ K,
    const _Float16* __restrict__ V, _Float16* __restrict__ CTX)
{
    const int n = blockIdx.x;
    const int h = threadIdx.x >> 5, l = threadIdx.x & 31;
    const size_t base = (size_t)n * 512;
    const float qa = (float)Q[base + h * 64 + l];
    const float qb = (float)Q[base + h * 64 + l + 32];
    float sc[8]; float mx = -1e30f;
#pragma unroll
    for (int g = 0; g < 8; ++g) {
        float p = qa * (float)K[base + g * 64 + l] + qb * (float)K[base + g * 64 + l + 32];
#pragma unroll
        for (int o = 16; o > 0; o >>= 1) p += __shfl_xor(p, o, 32);
        p *= 0.125f;
        sc[g] = p; mx = fmaxf(mx, p);
    }
    float den = 0.f;
#pragma unroll
    for (int g = 0; g < 8; ++g) { sc[g] = __expf(sc[g] - mx); den += sc[g]; }
    const float inv = 1.f / den;
    float ca = 0.f, cb = 0.f;
#pragma unroll
    for (int g = 0; g < 8; ++g) {
        ca += sc[g] * (float)V[base + g * 64 + l];
        cb += sc[g] * (float)V[base + g * 64 + l + 32];
    }
    CTX[base + h * 64 + l]      = (_Float16)(ca * inv);
    CTX[base + h * 64 + l + 32] = (_Float16)(cb * inv);
}

// LayerNorm over last dim 512 with optional residual: Y = LN(X (+R)) * g + b.
// grid = rows/8; block = 256 (wave per row); lane covers 16 strided dims.
__global__ __launch_bounds__(256) void ln_kernel(
    const _Float16* __restrict__ X, const _Float16* __restrict__ R,
    const float* __restrict__ g, const float* __restrict__ b,
    _Float16* __restrict__ Y)
{
    const int wave = threadIdx.x >> 5, lane = threadIdx.x & 31;
    const size_t row = (size_t)blockIdx.x * 8 + wave;
    const _Float16* x = X + row * 512;
    const _Float16* r = R ? R + row * 512 : nullptr;
    float xr[16]; float s = 0.f, s2 = 0.f;
#pragma unroll
    for (int i = 0; i < 16; ++i) {
        const int d = lane + i * 32;
        float v = (float)x[d] + (r ? (float)r[d] : 0.f);
        xr[i] = v; s += v; s2 += v * v;
    }
#pragma unroll
    for (int o = 16; o > 0; o >>= 1) { s += __shfl_xor(s, o, 32); s2 += __shfl_xor(s2, o, 32); }
    const float mu   = s * (1.f / 512.f);
    const float var  = s2 * (1.f / 512.f) - mu * mu;
    const float rstd = rsqrtf(var + 1e-5f);
#pragma unroll
    for (int i = 0; i < 16; ++i) {
        const int d = lane + i * 32;
        Y[row * 512 + d] = (_Float16)((xr[i] - mu) * rstd * g[d] + b[d]);
    }
}

// Gather frame_reps rows into shot layout: FT row m=((s*32+w)*128+l)
// comes from FR row (bounds[s,0]+l)*32 + w.  16-byte chunks.
__global__ void permute_ft(const _Float16* __restrict__ FR, const int* __restrict__ bounds,
                           _Float16* __restrict__ FT)
{
    const size_t idx = (size_t)blockIdx.x * 256 + threadIdx.x; // over 65536*64 uint4 chunks
    const int m = (int)(idx >> 6), c = (int)(idx & 63);
    const int l = m & 127, sw = m >> 7, w = sw & 31, s = sw >> 5;
    const int t = bounds[2 * s] + l;
    const uint4* src = (const uint4*)(FR + (size_t)(t * 32 + w) * 512);
    uint4* dst = (uint4*)(FT + (size_t)m * 512);
    dst[c] = src[c];
}

// Shot-level attention: one block per (s,l); thread = (head h, query row w).
// K tile staged in 32KB LDS; V streamed from global (L2-resident per block).
__global__ __launch_bounds__(256) void shot_attn(
    const _Float16* __restrict__ SQ, const _Float16* __restrict__ SK,
    const _Float16* __restrict__ SV, _Float16* __restrict__ SCTX)
{
    __shared__ alignas(16) _Float16 Ks[32 * 512];   // 32 KB
    const int s = blockIdx.x >> 7;
    const int l = blockIdx.x & 127;
    const int tid = threadIdx.x;
    {
        const int w = tid >> 3, part = tid & 7;
        const uint4* src = (const uint4*)(SK + ((size_t)(s * 32 + w) * 128 + l) * 512);
        uint4* dst = (uint4*)(Ks + w * 512);
#pragma unroll
        for (int i = 0; i < 8; ++i) dst[part * 8 + i] = src[part * 8 + i];
    }
    __syncthreads();
    const int h = tid >> 5, w = tid & 31;
    const size_t m = (size_t)(s * 32 + w) * 128 + l;
    const _Float16* q = SQ + m * 512 + h * 64;
    float qf[64];
#pragma unroll
    for (int d = 0; d < 64; ++d) qf[d] = (float)q[d];
    float sc[32]; float mx = -1e30f;
    for (int v = 0; v < 32; ++v) {
        float a = 0.f;
#pragma unroll
        for (int d = 0; d < 64; ++d) a += qf[d] * (float)Ks[v * 512 + h * 64 + d];
        a *= 0.125f;
        sc[v] = a; mx = fmaxf(mx, a);
    }
    float den = 0.f;
    for (int v = 0; v < 32; ++v) { sc[v] = __expf(sc[v] - mx); den += sc[v]; }
    const float inv = 1.f / den;
    for (int d = 0; d < 64; ++d) {
        float a = 0.f;
        for (int v = 0; v < 32; ++v) {
            const size_t mv = (size_t)(s * 32 + v) * 128 + l;
            a += sc[v] * (float)SV[mv * 512 + h * 64 + d];
        }
        SCTX[m * 512 + h * 64 + d] = (_Float16)(a * inv);
    }
}

// Row softmax over rows of 1024 f16 values, in place.  grid = 65536 rows.
__global__ __launch_bounds__(256) void row_softmax(_Float16* __restrict__ P)
{
    __shared__ float red[256];
    const size_t row = blockIdx.x;
    _Float16* p = P + row * 1024;
    const int tid = threadIdx.x;
    float v[4]; float mx = -1e30f;
#pragma unroll
    for (int i = 0; i < 4; ++i) { v[i] = (float)p[tid + i * 256]; mx = fmaxf(mx, v[i]); }
    red[tid] = mx; __syncthreads();
    for (int s2 = 128; s2 > 0; s2 >>= 1) { if (tid < s2) red[tid] = fmaxf(red[tid], red[tid + s2]); __syncthreads(); }
    mx = red[0]; __syncthreads();
    float sum = 0.f;
#pragma unroll
    for (int i = 0; i < 4; ++i) { v[i] = __expf(v[i] - mx); sum += v[i]; }
    red[tid] = sum; __syncthreads();
    for (int s2 = 128; s2 > 0; s2 >>= 1) { if (tid < s2) red[tid] += red[tid + s2]; __syncthreads(); }
    const float inv = 1.f / red[0];
#pragma unroll
    for (int i = 0; i < 4; ++i) p[tid + i * 256] = (_Float16)(v[i] * inv);
}

// logits[m] = SO[m,:] . pW[0:512] + ATT[m,:] . pW[512:1024] + pb
__global__ __launch_bounds__(256) void logits_kernel(
    const _Float16* __restrict__ SO, const _Float16* __restrict__ ATT,
    const float* __restrict__ pW, const float* __restrict__ pb,
    float* __restrict__ logits)
{
    const int wave = threadIdx.x >> 5, lane = threadIdx.x & 31;
    const size_t m = (size_t)blockIdx.x * 8 + wave;
    float a = 0.f;
#pragma unroll
    for (int i = 0; i < 16; ++i) {
        const int d = lane + i * 32;
        a += (float)SO[m * 512 + d] * pW[d] + (float)ATT[m * 512 + d] * pW[512 + d];
    }
#pragma unroll
    for (int o = 16; o > 0; o >>= 1) a += __shfl_xor(a, o, 32);
    if (lane == 0) logits[m] = a + pb[0];
}

// Softmax over axis 0 (SW=512 rows) for each column l (L=128). grid=128.
__global__ __launch_bounds__(256) void col_softmax(const float* __restrict__ logits,
                                                   float* __restrict__ out)
{
    __shared__ float red[256];
    const int l = blockIdx.x;
    const int tid = threadIdx.x;
    const float v0 = logits[(size_t)tid * 128 + l];
    const float v1 = logits[(size_t)(tid + 256) * 128 + l];
    red[tid] = fmaxf(v0, v1); __syncthreads();
    for (int s2 = 128; s2 > 0; s2 >>= 1) { if (tid < s2) red[tid] = fmaxf(red[tid], red[tid + s2]); __syncthreads(); }
    const float mx = red[0]; __syncthreads();
    const float e0 = __expf(v0 - mx), e1 = __expf(v1 - mx);
    red[tid] = e0 + e1; __syncthreads();
    for (int s2 = 128; s2 > 0; s2 >>= 1) { if (tid < s2) red[tid] += red[tid + s2]; __syncthreads(); }
    const float inv = 1.f / red[0];
    out[(size_t)tid * 128 + l] = e0 * inv;
    out[(size_t)(tid + 256) * 128 + l] = e1 * inv;
}

// ---------------------------------------------------------------------------
extern "C" void kernel_launch(void* const* d_in, const int* in_sizes, int n_in,
                              void* d_out, int out_size, void* d_ws, size_t ws_size,
                              hipStream_t stream)
{
    (void)in_sizes; (void)n_in; (void)out_size; (void)ws_size;

    const float* visual = (const float*)d_in[0];   // [2048,32,512]
    const float* audio  = (const float*)d_in[1];   // [1024,512]
    const int*   bounds = (const int*)d_in[2];     // [16,2]

    // bias / LN / final params (all f32, used directly)
    const float* f_bq = (const float*)d_in[4];
    const float* f_bk = (const float*)d_in[6];
    const float* f_bv = (const float*)d_in[8];
    const float* f_ln_g = (const float*)d_in[9];
    const float* f_ln_b = (const float*)d_in[10];
    const float* f_lin_b = (const float*)d_in[12];
    const float* f_b0 = (const float*)d_in[14];
    const float* s_bq = (const float*)d_in[16];
    const float* s_bk = (const float*)d_in[18];
    const float* s_bv = (const float*)d_in[20];
    const float* s_bo = (const float*)d_in[22];
    const float* s_ln1_g = (const float*)d_in[23];
    const float* s_ln1_b = (const float*)d_in[24];
    const float* s_ff1_b = (const float*)d_in[26];
    const float* s_ff2_b = (const float*)d_in[28];
    const float* s_ln2_g = (const float*)d_in[29];
    const float* s_ln2_b = (const float*)d_in[30];
    const float* a_bq = (const float*)d_in[32];
    const float* a_bk = (const float*)d_in[34];
    const float* a_bv = (const float*)d_in[36];
    const float* p_W = (const float*)d_in[37];
    const float* p_b = (const float*)d_in[38];

    // ---- workspace layout ----
    char* ws = (char*)d_ws;
    const size_t WSZ = 512 * 512;                        // halves per weight matrix
    _Float16* W16 = (_Float16*)ws;                       // 14 transposed f16 weights
    size_t off = 14 * WSZ * sizeof(_Float16);
    _Float16* audio16 = (_Float16*)(ws + off); off += (size_t)1024 * 512 * 2;
    _Float16* AK16    = (_Float16*)(ws + off); off += (size_t)1024 * 512 * 2;
    _Float16* AV16    = (_Float16*)(ws + off); off += (size_t)1024 * 512 * 2;
    _Float16* AVt16   = (_Float16*)(ws + off); off += (size_t)512 * 1024 * 2;
    float*    logitsB = (float*)(ws + off);    off += (size_t)65536 * 4;
    off = (off + 255) & ~(size_t)255;
    const size_t UNIT = (size_t)65536 * 512 * 2;         // 64 MB f16 slab
    _Float16* U0 = (_Float16*)(ws + off);
    _Float16* U1 = (_Float16*)(ws + off + 1 * UNIT);
    _Float16* U2 = (_Float16*)(ws + off + 2 * UNIT);
    _Float16* U3 = (_Float16*)(ws + off + 3 * UNIT);
    _Float16* U4 = (_Float16*)(ws + off + 4 * UNIT);     // U3+U4 contiguous -> ASC

    // weight slot order (all [512,512] -> transposed f16)
    const int widx[14] = {3, 5, 7, 11, 13, 15, 17, 19, 21, 25, 27, 31, 33, 35};
    enum { WQF, WKF, WVF, WLIN, W0F, SWQ, SWK, SWV, SWO, SF1, SF2, AWQ, AWK, AWV };
    for (int i = 0; i < 14; ++i)
        wprep_kernel<<<1024, 256, 0, stream>>>((const float*)d_in[widx[i]],
                                               W16 + (size_t)i * WSZ, 512, 512);

    auto gemm = [&](const _Float16* A, int wslot, const float* bias, _Float16* C,
                    int M, int N, int K, int relu, const _Float16* BtOverride = nullptr) {
        dim3 g(N / TILE_N, M / TILE_M);
        const _Float16* Bt = BtOverride ? BtOverride : (W16 + (size_t)wslot * WSZ);
        gemm_wmma_f16<<<g, 256, 0, stream>>>(A, Bt, bias, C, M, N, K, relu);
    };

    // ---- convert inputs to f16 ----
    f32_to_f16<<<(33554432 + 255) / 256, 256, 0, stream>>>(visual, U0, (size_t)33554432);
    f32_to_f16<<<2048, 256, 0, stream>>>(audio, audio16, (size_t)524288);

    // ---- stage 1: frame-level transformer ----
    gemm(U0, WQF, f_bq, U1, 65536, 512, 512, 0);           // Q
    gemm(U0, WKF, f_bk, U2, 65536, 512, 512, 0);           // K
    gemm(U0, WVF, f_bv, U3, 65536, 512, 512, 0);           // V
    frame_attn<<<65536, 256, 0, stream>>>(U1, U2, U3, U4); // ctx
    ln_kernel<<<8192, 256, 0, stream>>>(U4, nullptr, f_ln_g, f_ln_b, U0);
    gemm(U0, WLIN, f_lin_b, U1, 65536, 512, 512, 1);       // relu(LN @ lin)
    gemm(U1, W0F, f_b0, U2, 65536, 512, 512, 0);           // frame_reps

    // ---- stage 2: shot-level transformer ----
    permute_ft<<<16384, 256, 0, stream>>>(U2, bounds, U3); // ft
    gemm(U3, SWQ, s_bq, U0, 65536, 512, 512, 0);           // sq
    gemm(U3, SWK, s_bk, U1, 65536, 512, 512, 0);           // sk
    gemm(U3, SWV, s_bv, U4, 65536, 512, 512, 0);           // sv
    shot_attn<<<2048, 256, 0, stream>>>(U0, U1, U4, U2);   // sctx
    gemm(U2, SWO, s_bo, U0, 65536, 512, 512, 0);           // attn_out
    ln_kernel<<<8192, 256, 0, stream>>>(U3, U0, s_ln1_g, s_ln1_b, U1); // hh
    gemm(U1, SF1, s_ff1_b, U4, 65536, 512, 512, 1);        // relu(hh @ ff1)
    gemm(U4, SF2, s_ff2_b, U0, 65536, 512, 512, 0);        // ff
    ln_kernel<<<8192, 256, 0, stream>>>(U1, U0, s_ln2_g, s_ln2_b, U2); // SO = combined

    // ---- stage 3: audio-guided cross attention ----
    gemm(audio16, AWK, a_bk, AK16, 1024, 512, 512, 0);     // ak [A,D]
    gemm(audio16, AWV, a_bv, AV16, 1024, 512, 512, 0);     // av [A,D]
    transpose_f16<<<2048, 256, 0, stream>>>(AV16, AVt16, 1024, 512); // av^T [D,A]
    gemm(U2, AWQ, a_bq, U0, 65536, 512, 512, 0);           // aq
    _Float16* ASC = U3;                                    // [65536,1024] spans U3+U4
    gemm(U0, 0, nullptr, ASC, 65536, 1024, 512, 0, AK16);  // scores = aq @ ak^T
    row_softmax<<<65536, 256, 0, stream>>>(ASC);           // softmax over A
    gemm(ASC, 0, nullptr, U1, 65536, 512, 1024, 0, AVt16); // attended = P @ av

    // ---- final probability layer + softmax over SW axis ----
    logits_kernel<<<8192, 256, 0, stream>>>(U2, U1, p_W, p_b, logitsB);
    col_softmax<<<128, 256, 0, stream>>>(logitsB, (float*)d_out);
}
